// VisionMamba_40046275068522
// MI455X (gfx1250) — matrix operand order repaired and checked by reading, not compiled, not run on hardware
//
#include <hip/hip_runtime.h>

typedef __bf16 bf16;
typedef bf16  v16bf __attribute__((ext_vector_type(16)));
typedef bf16  v8bf  __attribute__((ext_vector_type(8)));
typedef float v8f   __attribute__((ext_vector_type(8)));

// ---- model constants ----
constexpr int cD = 768, cDI = 1536, cN = 16, cR = 48, cKC = 4, cDEPTH = 24;
constexpr int cB = 4, cH = 1280, cW = 960, cP = 64, cCIN = 3;
constexpr int cGH = cH / cP, cGW = cW / cP;      // 20, 15
constexpr int cNP = cGH * cGW;                   // 300
constexpr int cL  = cNP + 1;                     // 301
constexpr int cPOS = cNP / 2;                    // 150
constexpr int cNCLS = 1000;
constexpr int cXD = cR + 2 * cN;                 // 80
constexpr int cKP = cCIN * cP * cP;              // 12288
constexpr float cEPS = 1e-5f;

// =====================================================================
// WMMA GEMM:  C[m,n] = sum_k A[m*lda+k] * W[n*ldb+k]  (+ bias[n])
// A: MxK row-major f32, W: NxK row-major f32, C: MxN row-major f32.
// bf16 WMMA 16x16x32. Block tile 128x64, 8 waves (4M x 2N),
// wave tile 32x32 -> 4 v_wmma per k-step per wave.
// Ping-pong LDS double buffering: one barrier per k-step; global loads
// of tile t+1 overlap the WMMAs of tile t.
// M/N edges: row clamping (garbage rows feed only unstored C).
// K edge: uniform branch, zero-filled.
// =====================================================================
#define BM 128
#define BN 64
#define BK 32
#define LDSTR 40   // LDS row stride in bf16: 80B rows, 16B-aligned v8bf stores

__global__ __launch_bounds__(256) void wmma_gemm_nt(
    const float* __restrict__ A, int lda,
    const float* __restrict__ Wt, int ldb,
    const float* __restrict__ bias,
    float* __restrict__ C, int ldc,
    int M, int N, int K)
{
  __shared__ __attribute__((aligned(16))) bf16 lA[2][BM * LDSTR];
  __shared__ __attribute__((aligned(16))) bf16 lB[2][BN * LDSTR];

  const int tid  = threadIdx.x;
  const int wave = tid >> 5;
  const int lane = tid & 31;
  const int m0 = blockIdx.y * BM;
  const int n0 = blockIdx.x * BN;

  const int wm = (wave & 3) * 32;   // wave M offset in block
  const int wn = (wave >> 2) * 32;  // wave N offset in block

  const int h  = lane >> 4;         // lane half
  const int mr = lane & 15;         // row/col within 16

  v8f acc00 = {}, acc01 = {}, acc10 = {}, acc11 = {};

  // staging: A = 128 rows x 32 k, 2 threads/row (16 floats each)
  //          B =  64 rows x 32 k, 4 threads/row ( 8 floats each)
  const int arow = tid >> 1;
  const int acol = (tid & 1) * 16;
  const int brow = tid >> 2;
  const int bcol = (tid & 3) * 8;

  const int gmc = (m0 + arow < M) ? (m0 + arow) : (M - 1);  // clamped row
  const int gnc = (n0 + brow < N) ? (n0 + brow) : (N - 1);
  const float* __restrict__ Ap = A  + (size_t)gmc * lda;
  const float* __restrict__ Bp = Wt + (size_t)gnc * ldb;

  float4 ra0, ra1, ra2, ra3, rb0, rb1;   // staging registers (in flight)

  auto load_tile = [&](int k0) {
    if (k0 + BK <= K) {                  // fast path, uniform condition
      ra0 = *(const float4*)(Ap + k0 + acol);
      ra1 = *(const float4*)(Ap + k0 + acol + 4);
      ra2 = *(const float4*)(Ap + k0 + acol + 8);
      ra3 = *(const float4*)(Ap + k0 + acol + 12);
      rb0 = *(const float4*)(Bp + k0 + bcol);
      rb1 = *(const float4*)(Bp + k0 + bcol + 4);
    } else {                             // K edge: clamp + zero select
      float ta[16], tb[8];
#pragma unroll
      for (int i = 0; i < 16; ++i) {
        const int gk = k0 + acol + i;
        const int gkc = (gk < K) ? gk : (K - 1);
        const float v = Ap[gkc];
        ta[i] = (gk < K) ? v : 0.f;
      }
#pragma unroll
      for (int i = 0; i < 8; ++i) {
        const int gk = k0 + bcol + i;
        const int gkc = (gk < K) ? gk : (K - 1);
        const float v = Bp[gkc];
        tb[i] = (gk < K) ? v : 0.f;
      }
      ra0 = make_float4(ta[0], ta[1], ta[2], ta[3]);
      ra1 = make_float4(ta[4], ta[5], ta[6], ta[7]);
      ra2 = make_float4(ta[8], ta[9], ta[10], ta[11]);
      ra3 = make_float4(ta[12], ta[13], ta[14], ta[15]);
      rb0 = make_float4(tb[0], tb[1], tb[2], tb[3]);
      rb1 = make_float4(tb[4], tb[5], tb[6], tb[7]);
    }
  };

  load_tile(0);
  const int nk = (K + BK - 1) / BK;
  int buf = 0;

  for (int t = 0; t < nk; ++t) {
    // ---- commit staged tile to LDS[buf] ----
    {
      v8bf pa0 = { (bf16)ra0.x, (bf16)ra0.y, (bf16)ra0.z, (bf16)ra0.w,
                   (bf16)ra1.x, (bf16)ra1.y, (bf16)ra1.z, (bf16)ra1.w };
      v8bf pa1 = { (bf16)ra2.x, (bf16)ra2.y, (bf16)ra2.z, (bf16)ra2.w,
                   (bf16)ra3.x, (bf16)ra3.y, (bf16)ra3.z, (bf16)ra3.w };
      v8bf pb  = { (bf16)rb0.x, (bf16)rb0.y, (bf16)rb0.z, (bf16)rb0.w,
                   (bf16)rb1.x, (bf16)rb1.y, (bf16)rb1.z, (bf16)rb1.w };
      *(v8bf*)&lA[buf][arow * LDSTR + acol]     = pa0;
      *(v8bf*)&lA[buf][arow * LDSTR + acol + 8] = pa1;
      *(v8bf*)&lB[buf][brow * LDSTR + bcol]     = pb;
    }
    __syncthreads();

    // ---- kick off global loads for next tile (overlaps WMMA below) ----
    if (t + 1 < nk) load_tile((t + 1) * BK);

    // ---- gather fragments per ISA 7.12.2 16-bit layouts ----
    union U { unsigned u[8]; v16bf v; };
    U a0f, a1f, b0f, b1f;
#pragma unroll
    for (int vv = 0; vv < 8; ++vv) {
      const int kbA = (vv & 3) * 2 + (vv >> 2) * 16 + h * 8; // A packing
      const int kbB = (vv & 3) * 2 + (vv >> 2) * 16 + h * 8; // B packing
      a0f.u[vv] = *(const unsigned*)&lA[buf][(wm + mr) * LDSTR + kbA];
      a1f.u[vv] = *(const unsigned*)&lA[buf][(wm + 16 + mr) * LDSTR + kbA];
      b0f.u[vv] = *(const unsigned*)&lB[buf][(wn + mr) * LDSTR + kbB];
      b1f.u[vv] = *(const unsigned*)&lB[buf][(wn + 16 + mr) * LDSTR + kbB];
    }

    acc00 = __builtin_amdgcn_wmma_f32_16x16x32_bf16(false, a0f.v, false, b0f.v,
                                                    (short)0, acc00, false, false);
    acc01 = __builtin_amdgcn_wmma_f32_16x16x32_bf16(false, a0f.v, false, b1f.v,
                                                    (short)0, acc01, false, false);
    acc10 = __builtin_amdgcn_wmma_f32_16x16x32_bf16(false, a1f.v, false, b0f.v,
                                                    (short)0, acc10, false, false);
    acc11 = __builtin_amdgcn_wmma_f32_16x16x32_bf16(false, a1f.v, false, b1f.v,
                                                    (short)0, acc11, false, false);
    buf ^= 1;
  }

  // ---- store 4 subtiles: lane col = mr, rows r + 8*h ----
#pragma unroll
  for (int r = 0; r < 8; ++r) {
    const int rm0 = m0 + wm + r + 8 * h;
    const int rm1 = rm0 + 16;
    const int cn0 = n0 + wn + mr;
    const int cn1 = cn0 + 16;
    const float bb0 = (bias && cn0 < N) ? bias[cn0] : 0.f;
    const float bb1 = (bias && cn1 < N) ? bias[cn1] : 0.f;
    if (rm0 < M) {
      if (cn0 < N) C[(size_t)rm0 * ldc + cn0] = acc00[r] + bb0;
      if (cn1 < N) C[(size_t)rm0 * ldc + cn1] = acc01[r] + bb1;
    }
    if (rm1 < M) {
      if (cn0 < N) C[(size_t)rm1 * ldc + cn0] = acc10[r] + bb0;
      if (cn1 < N) C[(size_t)rm1 * ldc + cn1] = acc11[r] + bb1;
    }
  }
}

// =====================================================================
// im2row (float4): img (B,CIN,H,W) -> patches (B*NP, 12288)
// =====================================================================
__global__ void im2row_kernel(const float* __restrict__ img, float4* __restrict__ patches, long n4)
{
  long idx = (long)blockIdx.x * blockDim.x + threadIdx.x;
  if (idx >= n4) return;
  const int KP4 = cKP / 4;
  int k4 = (int)(idx % KP4);
  long m = idx / KP4;
  int j = (int)(m % cGW);
  int i = (int)((m / cGW) % cGH);
  int b = (int)(m / ((long)cGW * cGH));
  int q = (k4 * 4) % cP;
  int p = (k4 * 4 / cP) % cP;
  int c = (k4 * 4) / (cP * cP);
  patches[idx] = *(const float4*)&img[(((size_t)b * cCIN + c) * cH + i * cP + p) * cW + j * cP + q];
}

// =====================================================================
// assemble tokens: insert cls at POS, add pos_embed; zero residual
// =====================================================================
__global__ void assemble_kernel(const float* __restrict__ pe, const float* __restrict__ cls,
                                const float* __restrict__ pos, float* __restrict__ hbuf,
                                float* __restrict__ res, long n)
{
  long idx = (long)blockIdx.x * blockDim.x + threadIdx.x;
  if (idx >= n) return;
  int d = (int)(idx % cD);
  int l = (int)((idx / cD) % cL);
  int b = (int)(idx / ((long)cD * cL));
  float v;
  if (l == cPOS) v = cls[d];
  else {
    int pi = (l < cPOS) ? l : l - 1;
    v = pe[((size_t)b * cNP + pi) * cD + d];
  }
  hbuf[idx] = v + pos[(size_t)l * cD + d];
  res[idx] = 0.f;
}

// =====================================================================
// fused residual + RMSNorm: res += h; hn = res * rsqrt(mean(res^2)+eps) * w
// =====================================================================
__global__ __launch_bounds__(256) void rmsnorm_residual_kernel(
    const float* __restrict__ h, float* __restrict__ res,
    const float* __restrict__ w, float* __restrict__ hn)
{
  const int t = blockIdx.x;
  const int tx = threadIdx.x;
  __shared__ float red[256];
  float vals[3];
  float local = 0.f;
#pragma unroll
  for (int i = 0; i < 3; ++i) {
    const int d = tx + i * 256;
    float v = res[(size_t)t * cD + d] + h[(size_t)t * cD + d];
    vals[i] = v;
    local += v * v;
  }
  red[tx] = local;
  __syncthreads();
  for (int s = 128; s > 0; s >>= 1) {
    if (tx < s) red[tx] += red[tx + s];
    __syncthreads();
  }
  const float inv = rsqrtf(red[0] / (float)cD + cEPS);
#pragma unroll
  for (int i = 0; i < 3; ++i) {
    const int d = tx + i * 256;
    res[(size_t)t * cD + d] = vals[i];
    hn[(size_t)t * cD + d] = vals[i] * inv * w[d];
  }
}

// =====================================================================
// causal depthwise conv (K=4) + SiLU, optional sequence reversal
// =====================================================================
__global__ void conv_silu_kernel(const float* __restrict__ xz, const float* __restrict__ w,
                                 const float* __restrict__ bias, float* __restrict__ xc,
                                 int reverse, long n)
{
  long idx = (long)blockIdx.x * blockDim.x + threadIdx.x;
  if (idx >= n) return;
  int d = (int)(idx % cDI);
  long bl = idx / cDI;
  int l = (int)(bl % cL);
  int b = (int)(bl / cL);
  float acc = bias[d];
#pragma unroll
  for (int k = 0; k < cKC; ++k) {
    const int p = l - (cKC - 1) + k;
    if (p >= 0) {
      const int s = reverse ? (cL - 1 - p) : p;
      acc += w[d * cKC + k] * xz[((size_t)b * cL + s) * (2 * cDI) + d];
    }
  }
  xc[idx] = acc / (1.f + __expf(-acc));   // SiLU
}

// =====================================================================
// softplus(dtraw + dt_b)
// =====================================================================
__global__ void softplus_kernel(const float* __restrict__ dtraw, const float* __restrict__ dtb,
                                float* __restrict__ dt, long n)
{
  long idx = (long)blockIdx.x * blockDim.x + threadIdx.x;
  if (idx >= n) return;
  int d = (int)(idx % cDI);
  float x = dtraw[idx] + dtb[d];
  dt[idx] = (x > 20.f) ? x : log1pf(__expf(x));
}

// =====================================================================
// selective scan: block-uniform b so the shared (B,C) row loads are
// wave-uniform (SMEM-eligible); one thread per d, 16 states in regs.
// =====================================================================
constexpr int cDBLK = cDI / 256;   // 6 blocks per batch element

__global__ __launch_bounds__(256) void ssm_scan_kernel(
    const float* __restrict__ xc, const float* __restrict__ zbase, int zstride,
    const float* __restrict__ dbl, const float* __restrict__ dt,
    const float* __restrict__ A_log, const float* __restrict__ Dp,
    float* __restrict__ ybuf, int reverse)
{
  const int b = blockIdx.x / cDBLK;                       // scalar
  const int d = (blockIdx.x % cDBLK) * 256 + threadIdx.x; // vector
  float a[cN], hst[cN];
#pragma unroll
  for (int nn = 0; nn < cN; ++nn) {
    a[nn] = -__expf(A_log[(size_t)d * cN + nn]);
    hst[nn] = 0.f;
  }
  const float dval = Dp[d];
  for (int l = 0; l < cL; ++l) {
    const size_t bl = (size_t)b * cL + l;
    const float dtv = dt[bl * cDI + d];
    const float xcv = xc[bl * cDI + d];
    const float du = dtv * xcv;
    const float* __restrict__ row = dbl + bl * cXD;       // uniform address
    float y = 0.f;
#pragma unroll
    for (int nn = 0; nn < cN; ++nn) {
      float hv = __expf(dtv * a[nn]) * hst[nn] + du * row[cR + nn];
      hst[nn] = hv;
      y += hv * row[cR + cN + nn];
    }
    y += xcv * dval;
    const int ls = reverse ? (cL - 1 - l) : l;
    const float zv = zbase[((size_t)b * cL + ls) * zstride + d];
    const float yo = 0.5f * y * zv / (1.f + __expf(-zv));
    const size_t oi = ((size_t)b * cL + ls) * cDI + d;
    if (reverse) ybuf[oi] += yo;
    else         ybuf[oi]  = yo;
  }
}

// =====================================================================
// final: v = res[b,POS]+h[b,POS]; RMS-norm with norm_f_w -> outpos (B x D)
// =====================================================================
__global__ __launch_bounds__(256) void final_rms_kernel(
    const float* __restrict__ h, const float* __restrict__ res,
    const float* __restrict__ w, float* __restrict__ outpos)
{
  const int b = blockIdx.x;
  const int tx = threadIdx.x;
  const size_t base = ((size_t)b * cL + cPOS) * cD;
  __shared__ float red[256];
  float vals[3];
  float local = 0.f;
#pragma unroll
  for (int i = 0; i < 3; ++i) {
    const int d = tx + i * 256;
    float v = res[base + d] + h[base + d];
    vals[i] = v;
    local += v * v;
  }
  red[tx] = local;
  __syncthreads();
  for (int s = 128; s > 0; s >>= 1) {
    if (tx < s) red[tx] += red[tx + s];
    __syncthreads();
  }
  const float inv = rsqrtf(red[0] / (float)cD + cEPS);
#pragma unroll
  for (int i = 0; i < 3; ++i) {
    const int d = tx + i * 256;
    outpos[(size_t)b * cD + d] = vals[i] * inv * w[d];
  }
}

// =====================================================================
// host orchestration
// =====================================================================
static inline dim3 gemm_grid(int M, int N) {
  return dim3((unsigned)((N + BN - 1) / BN), (unsigned)((M + BM - 1) / BM));
}

extern "C" void kernel_launch(void* const* d_in, const int* in_sizes, int n_in,
                              void* d_out, int out_size, void* d_ws, size_t ws_size,
                              hipStream_t stream) {
  const float* img       = (const float*)d_in[0];
  const float* patch_w   = (const float*)d_in[1];
  const float* patch_b   = (const float*)d_in[2];
  const float* cls_token = (const float*)d_in[3];
  const float* pos_embed = (const float*)d_in[4];
  const float* norm_w    = (const float*)d_in[5];
  const float* in_proj_w = (const float*)d_in[6];
  const float* conv_w[2]  = { (const float*)d_in[7],  (const float*)d_in[14] };
  const float* conv_b[2]  = { (const float*)d_in[8],  (const float*)d_in[15] };
  const float* xproj_w[2] = { (const float*)d_in[9],  (const float*)d_in[16] };
  const float* dt_w[2]    = { (const float*)d_in[10], (const float*)d_in[17] };
  const float* dt_b[2]    = { (const float*)d_in[11], (const float*)d_in[18] };
  const float* A_log[2]   = { (const float*)d_in[12], (const float*)d_in[19] };
  const float* Dp[2]      = { (const float*)d_in[13], (const float*)d_in[20] };
  const float* out_proj_w = (const float*)d_in[21];
  const float* norm_f_w   = (const float*)d_in[22];
  const float* head_w     = (const float*)d_in[23];
  const float* head_b     = (const float*)d_in[24];
  float* out = (float*)d_out;

  // ---- workspace carve-up ----
  char* ws = (char*)d_ws;
  size_t off = 0;
  auto alloc = [&](size_t nfloats) -> float* {
    float* p = (float*)(ws + off);
    off += ((nfloats * sizeof(float) + 255) / 256) * 256;
    return p;
  };
  const int M = cB * cL;                                   // 1204
  float* patches = alloc((size_t)cB * cNP * cKP);
  float* pe      = alloc((size_t)cB * cNP * cD);
  float* hbuf    = alloc((size_t)M * cD);
  float* resb    = alloc((size_t)M * cD);
  float* hn      = alloc((size_t)M * cD);
  float* xz      = alloc((size_t)M * 2 * cDI);
  float* xcb     = alloc((size_t)M * cDI);
  float* dblb    = alloc((size_t)M * cXD);
  float* dtraw   = alloc((size_t)M * cDI);
  float* dtb     = alloc((size_t)M * cDI);
  float* ybuf    = alloc((size_t)M * cDI);
  float* outpos  = alloc((size_t)cB * cD);
  (void)ws_size; (void)n_in; (void)in_sizes; (void)out_size;

  const dim3 blk(256);
  auto grid1d = [](long n) { return dim3((unsigned)((n + 255) / 256)); };

  // ---- patch embedding ----
  const long nim4 = (long)cB * cNP * cKP / 4;
  im2row_kernel<<<grid1d(nim4), blk, 0, stream>>>(img, (float4*)patches, nim4);
  wmma_gemm_nt<<<gemm_grid(cB * cNP, cD), blk, 0, stream>>>(
      patches, cKP, patch_w, cKP, patch_b, pe, cD, cB * cNP, cD, cKP);

  const long ntok = (long)M * cD;
  assemble_kernel<<<grid1d(ntok), blk, 0, stream>>>(pe, cls_token, pos_embed, hbuf, resb, ntok);

  const long nch = (long)M * cDI;

  // ---- layers ----
  for (int lyr = 0; lyr < cDEPTH; ++lyr) {
    rmsnorm_residual_kernel<<<dim3((unsigned)M), blk, 0, stream>>>(
        hbuf, resb, norm_w + (size_t)lyr * cD, hn);

    wmma_gemm_nt<<<gemm_grid(M, 2 * cDI), blk, 0, stream>>>(
        hn, cD, in_proj_w + (size_t)lyr * 2 * cDI * cD, cD, nullptr,
        xz, 2 * cDI, M, 2 * cDI, cD);

    for (int dir = 0; dir < 2; ++dir) {
      conv_silu_kernel<<<grid1d(nch), blk, 0, stream>>>(
          xz, conv_w[dir] + (size_t)lyr * cDI * cKC, conv_b[dir] + (size_t)lyr * cDI,
          xcb, dir, nch);

      wmma_gemm_nt<<<gemm_grid(M, cXD), blk, 0, stream>>>(
          xcb, cDI, xproj_w[dir] + (size_t)lyr * cXD * cDI, cDI, nullptr,
          dblb, cXD, M, cXD, cDI);

      wmma_gemm_nt<<<gemm_grid(M, cDI), blk, 0, stream>>>(
          dblb, cXD, dt_w[dir] + (size_t)lyr * cDI * cR, cR, nullptr,
          dtraw, cDI, M, cDI, cR);

      softplus_kernel<<<grid1d(nch), blk, 0, stream>>>(
          dtraw, dt_b[dir] + (size_t)lyr * cDI, dtb, nch);

      ssm_scan_kernel<<<dim3((unsigned)(cB * cDBLK)), blk, 0, stream>>>(
          xcb, xz + cDI, 2 * cDI, dblb, dtb,
          A_log[dir] + (size_t)lyr * cDI * cN, Dp[dir] + (size_t)lyr * cDI,
          ybuf, dir);
    }

    wmma_gemm_nt<<<gemm_grid(M, cD), blk, 0, stream>>>(
        ybuf, cDI, out_proj_w + (size_t)lyr * cD * cDI, cDI, nullptr,
        hbuf, cD, M, cD, cDI);
  }

  // ---- head ----
  final_rms_kernel<<<dim3((unsigned)cB), blk, 0, stream>>>(hbuf, resb, norm_f_w, outpos);
  wmma_gemm_nt<<<gemm_grid(cB, cNCLS), blk, 0, stream>>>(
      outpos, cD, head_w, cD, head_b, out, cNCLS, cB, cNCLS, cD);
}
